// SpaMSLA_71399536328726
// MI455X (gfx1250) — matrix operand order repaired and checked by reading, not compiled, not run on hardware
//
#include <hip/hip_runtime.h>
#include <hip/hip_bf16.h>

// ---------------------------------------------------------------------------
// SpaMSLA for MI455X (gfx1250): all 20000-row dense GEMMs on bf16 WMMA
// (v_wmma_f32_16x16x32_bf16, f32 accumulate), everything else on VALU.
// GEMM v2: vectorized fast-path staging (float4 -> packed b128 LDS stores),
// clamped branch-free edge path, b128 LDS fragment loads, 16x32 per wave.
// ---------------------------------------------------------------------------

typedef __attribute__((ext_vector_type(16))) __bf16 v16bf;
typedef __attribute__((ext_vector_type(8)))  __bf16 v8bf;
typedef __attribute__((ext_vector_type(8)))  float  v8f;

#define NR    20000   // nodes
#define IN1F  3000
#define IN2F  256
#define HIDN  128
#define OUTF  64
#define LNUM  3
#define MFEAT 30      // FAVOR+ random features
#define KSAMP 10      // gumbel samples
#define EPSC  1e-8f

#define BM 64
#define BN 64
#define BKT 32
#define LDP 40        // padded LDS row stride (bf16 units), 80B keeps 16B align

__device__ __forceinline__ __bf16 f2bf(float x) {
  unsigned u = __builtin_bit_cast(unsigned, x);
  unsigned r = (u + 0x7FFFu + ((u >> 16) & 1u)) >> 16;
  return __builtin_bit_cast(__bf16, (unsigned short)r);
}

// ------------------------- WMMA GEMM: C = A*B (+bias) ----------------------
// A [Md x Kd] fp32 row-major (lda), B [Kd x Nd] fp32 row-major (ldb),
// C [Md x Nd] fp32 (ldc). fp32 -> bf16 conversion happens in the LDS loaders.
// Block: 256 threads = 8 waves; block tile 64x64; wave tile 16x32 (2 WMMAs).
__global__ __launch_bounds__(256) void gemm_wmma_bf16(
    const float* __restrict__ A, int lda,
    const float* __restrict__ B, int ldb,
    const float* __restrict__ bias,
    float* __restrict__ C, int ldc,
    int Md, int Nd, int Kd) {
  __shared__ __align__(16) __bf16 As[BM][LDP];   // row-major [m][k]
  __shared__ __align__(16) __bf16 Bs[BN][LDP];   // transposed [n][k]

  const int tid  = threadIdx.x;
  const int bm   = blockIdx.y * BM;
  const int bn   = blockIdx.x * BN;
  const int wave = tid >> 5;
  const int lane = tid & 31;
  const int wm   = (wave >> 1) * 16;   // 0,16,32,48
  const int wn   = (wave & 1) * 32;    // 0,32
  const int l15  = lane & 15;
  const int lhi  = lane >> 4;

  v8f acc0 = {}, acc1 = {};

  for (int kt = 0; kt < Kd; kt += BKT) {
    // ---- stage A tile 64x32: 8 floats per thread ----
    {
      int r = tid >> 2;               // 0..63
      int c = (tid & 3) * 8;          // 0,8,16,24
      if (bm + BM <= Md && kt + BKT <= Kd) {
        const float* ap = &A[(size_t)(bm + r) * lda + kt + c];
        float4 fa = *(const float4*)ap;
        float4 fb = *(const float4*)(ap + 4);
        v8bf pk;
        pk[0] = f2bf(fa.x); pk[1] = f2bf(fa.y); pk[2] = f2bf(fa.z); pk[3] = f2bf(fa.w);
        pk[4] = f2bf(fb.x); pk[5] = f2bf(fb.y); pk[6] = f2bf(fb.z); pk[7] = f2bf(fb.w);
        *(v8bf*)&As[r][c] = pk;
      } else {
        int gr = bm + r; int grc = gr < Md ? gr : Md - 1;
#pragma unroll
        for (int j = 0; j < 8; ++j) {
          int gc = kt + c + j; int gcc = gc < Kd ? gc : Kd - 1;
          float v = A[(size_t)grc * lda + gcc];       // clamped: no exec branch
          As[r][c + j] = f2bf((gr < Md && gc < Kd) ? v : 0.f);
        }
      }
    }
    // ---- stage B tile 32x64 transposed into Bs[n][k]: 8 floats per thread ----
    {
      int r = tid >> 3;               // k row 0..31
      int c = (tid & 7) * 8;          // n col 0,8,..,56
      if (bn + BN <= Nd && kt + BKT <= Kd) {
        const float* bp = &B[(size_t)(kt + r) * ldb + bn + c];
        float4 fa = *(const float4*)bp;
        float4 fb = *(const float4*)(bp + 4);
        __bf16 pk[8];
        pk[0] = f2bf(fa.x); pk[1] = f2bf(fa.y); pk[2] = f2bf(fa.z); pk[3] = f2bf(fa.w);
        pk[4] = f2bf(fb.x); pk[5] = f2bf(fb.y); pk[6] = f2bf(fb.z); pk[7] = f2bf(fb.w);
#pragma unroll
        for (int j = 0; j < 8; ++j) Bs[c + j][r] = pk[j];
      } else {
        int gr = kt + r; int grc = gr < Kd ? gr : Kd - 1;
#pragma unroll
        for (int j = 0; j < 8; ++j) {
          int gc = bn + c + j; int gcc = gc < Nd ? gc : Nd - 1;
          float v = B[(size_t)grc * ldb + gcc];       // clamped: no exec branch
          Bs[c + j][r] = f2bf((gr < Kd && gc < Nd) ? v : 0.f);
        }
      }
    }
    // prefetch next A tile into GL2 (global_prefetch_b8)
    if (kt + BKT < Kd) {
      int pr = bm + (tid >> 2);
      if (pr < Md) __builtin_prefetch(&A[(size_t)pr * lda + kt + BKT], 0, 1);
    }
    __syncthreads();

    // ---- A fragment: two contiguous 16B runs of the LDS row (ds_load_b128) ----
    v8bf a0 = *(const v8bf*)&As[wm + l15][lhi * 8];
    v8bf a1 = *(const v8bf*)&As[wm + l15][16 + lhi * 8];
    v16bf afrag = __builtin_shufflevector(a0, a1,
        0, 1, 2, 3, 4, 5, 6, 7, 8, 9, 10, 11, 12, 13, 14, 15);
    // ---- B fragments: two contiguous 16B runs per subtile, same k order as A (ds_load_b128) ----
    v8bf b00 = *(const v8bf*)&Bs[wn + l15][lhi * 8];
    v8bf b01 = *(const v8bf*)&Bs[wn + l15][16 + lhi * 8];
    v16bf bfrag0 = __builtin_shufflevector(b00, b01,
        0, 1, 2, 3, 4, 5, 6, 7, 8, 9, 10, 11, 12, 13, 14, 15);
    v8bf b10 = *(const v8bf*)&Bs[wn + 16 + l15][lhi * 8];
    v8bf b11 = *(const v8bf*)&Bs[wn + 16 + l15][16 + lhi * 8];
    v16bf bfrag1 = __builtin_shufflevector(b10, b11,
        0, 1, 2, 3, 4, 5, 6, 7, 8, 9, 10, 11, 12, 13, 14, 15);

    acc0 = __builtin_amdgcn_wmma_f32_16x16x32_bf16(
        false, afrag, false, bfrag0, (short)0, acc0, false, false);
    acc1 = __builtin_amdgcn_wmma_f32_16x16x32_bf16(
        false, afrag, false, bfrag1, (short)0, acc1, false, false);
    __syncthreads();
  }

  // C store: lane -> N=l15 (+subtile), VGPR i -> M = i + lhi*8
#pragma unroll
  for (int i = 0; i < 8; ++i) {
    int r = bm + wm + i + lhi * 8;
    if (r < Md) {
      int c0 = bn + wn + l15;
      int c1 = c0 + 16;
      if (c0 < Nd) {
        float v = acc0[i];
        if (bias) v += bias[c0];
        C[(size_t)r * ldc + c0] = v;
      }
      if (c1 < Nd) {
        float v = acc1[i];
        if (bias) v += bias[c1];
        C[(size_t)r * ldc + c1] = v;
      }
    }
  }
}

// ------------------------- elementwise / reductions ------------------------

__global__ void fill_kernel(float* __restrict__ p, float v, size_t n) {
  size_t i = (size_t)blockIdx.x * blockDim.x + threadIdx.x;
  if (i < n) p[i] = v;
}

// LayerNorm (+optional residual) then ELU over rows of 128.
__global__ __launch_bounds__(128) void ln_elu_kernel(
    const float* __restrict__ x, int ldx,
    const float* __restrict__ resid, int ldr,
    const float* __restrict__ g, const float* __restrict__ b,
    float* __restrict__ out, int ldo) {
  __shared__ float red[128];
  int n = blockIdx.x, t = threadIdx.x;
  float v = x[(size_t)n * ldx + t];
  if (resid) v += resid[(size_t)n * ldr + t];
  red[t] = v;
  __syncthreads();
  for (int o = 64; o > 0; o >>= 1) {
    if (t < o) red[t] += red[t + o];
    __syncthreads();
  }
  float mu = red[0] * (1.f / 128.f);
  __syncthreads();
  float dv = v - mu;
  red[t] = dv * dv;
  __syncthreads();
  for (int o = 64; o > 0; o >>= 1) {
    if (t < o) red[t] += red[t + o];
    __syncthreads();
  }
  float var = red[0] * (1.f / 128.f);
  float y = dv * __frsqrt_rn(var + 1e-5f) * g[t] + b[t];
  out[(size_t)n * ldo + t] = (y > 0.f) ? y : (expf(y) - 1.f);
}

// FAVOR+ pass 1: dd[n,m] = <scale*q[n], P[m]>, diag, per-row max.
__global__ __launch_bounds__(32) void favor_dd_kernel(
    const float* __restrict__ q, int ldq, const float* __restrict__ P,
    float* __restrict__ dd, float* __restrict__ diag,
    float* __restrict__ rowmax, float scale) {
  __shared__ float row[HIDN];
  __shared__ float mx[32];
  int n = blockIdx.x, t = threadIdx.x;
  for (int i = t; i < HIDN; i += 32) row[i] = q[(size_t)n * ldq + i] * scale;
  __syncthreads();
  float d = 0.f;
  if (t < MFEAT) {
    for (int j = 0; j < HIDN; ++j) d += row[j] * P[t * HIDN + j];
    dd[(size_t)n * MFEAT + t] = d;
  } else if (t == 31) {
    float s = 0.f;
    for (int j = 0; j < HIDN; ++j) s += row[j] * row[j];
    diag[n] = 0.5f * s;
  }
  mx[t] = (t < MFEAT) ? d : -1e30f;
  __syncthreads();
  for (int o = 16; o > 0; o >>= 1) {
    if (t < o) mx[t] = fmaxf(mx[t], mx[t + o]);
    __syncthreads();
  }
  if (t == 0) rowmax[n] = mx[0];
}

__global__ __launch_bounds__(256) void reduce_max_kernel(
    const float* __restrict__ x, float* __restrict__ out, int n) {
  __shared__ float red[256];
  float m = -1e30f;
  for (int i = threadIdx.x; i < n; i += 256) m = fmaxf(m, x[i]);
  red[threadIdx.x] = m;
  __syncthreads();
  for (int o = 128; o > 0; o >>= 1) {
    if (threadIdx.x < o) red[threadIdx.x] = fmaxf(red[threadIdx.x], red[threadIdx.x + o]);
    __syncthreads();
  }
  if (threadIdx.x == 0) out[0] = red[0];
}

// FAVOR+ pass 2: phi = M^-0.5 * (exp(dd - diag - stab) + 1e-6), in place.
__global__ void favor_exp_kernel(float* __restrict__ dd,
                                 const float* __restrict__ diag,
                                 const float* __restrict__ rowmax,
                                 const float* __restrict__ gmax,
                                 int is_query, int total) {
  int i = blockIdx.x * blockDim.x + threadIdx.x;
  if (i >= total) return;
  int n = i / MFEAT;
  float stab = is_query ? rowmax[n] : gmax[0];
  dd[i] = 0.18257418583505536f * (expf(dd[i] - diag[n] - stab) + 1e-6f);
}

// kvs[k,m,d] = sum_n kp[n,m]*exp(gum[n,k])*v[n,d]; kgsum[k,m] = sum_n kp*eg.
__global__ __launch_bounds__(128) void kvs_kernel(
    const float* __restrict__ kp, const float* __restrict__ gum,
    const float* __restrict__ v, int ldv,
    float* __restrict__ kvs, float* __restrict__ kgsum, int n_rows) {
  int km = blockIdx.x;                 // 0..299
  int kk = km / MFEAT, m = km % MFEAT;
  int nsplit = gridDim.y;
  int per = (n_rows + nsplit - 1) / nsplit;
  int n0 = blockIdx.y * per;
  int n1 = min(n_rows, n0 + per);
  int d = threadIdx.x;
  float acc = 0.f, ks = 0.f;
  for (int n = n0; n < n1; ++n) {
    float w = kp[(size_t)n * MFEAT + m] * expf(gum[(size_t)n * KSAMP + kk]);
    acc += w * v[(size_t)n * ldv + d];
    ks += w;
  }
  atomicAdd(&kvs[(size_t)km * HIDN + d], acc);
  if (d == 0) atomicAdd(&kgsum[km], ks);
}

// zn[n,d] = mean_k [ (sum_m qp[n,m]*kvs[k,m,d]) / (sum_m qp[n,m]*kgsum[k,m]) ]
__global__ __launch_bounds__(128) void zn_kernel(
    const float* __restrict__ qp, const float* __restrict__ kvs,
    const float* __restrict__ kgsum, float* __restrict__ zn) {
  __shared__ float qs[MFEAT];
  __shared__ float den[KSAMP];
  int n = blockIdx.x, d = threadIdx.x;
  if (d < MFEAT) qs[d] = qp[(size_t)n * MFEAT + d];
  __syncthreads();
  if (d < KSAMP) {
    float s = 0.f;
    for (int m = 0; m < MFEAT; ++m) s += qs[m] * kgsum[d * MFEAT + m];
    den[d] = s;
  }
  __syncthreads();
  float acc = 0.f;
  for (int k = 0; k < KSAMP; ++k) {
    float num = 0.f;
    for (int m = 0; m < MFEAT; ++m) num += qs[m] * kvs[((size_t)k * MFEAT + m) * HIDN + d];
    acc += num / den[k];
  }
  zn[(size_t)n * HIDN + d] = acc * (1.f / (float)KSAMP);
}

// ------------------------- GCN / fusion ------------------------------------

__global__ void deg_edge_kernel(const int* __restrict__ dst, float* __restrict__ deg, int E) {
  int e = blockIdx.x * blockDim.x + threadIdx.x;
  if (e < E) atomicAdd(&deg[dst[e]], 1.f);
}

__global__ void dinv_kernel(const float* __restrict__ deg, float* __restrict__ dinv, int n) {
  int i = blockIdx.x * blockDim.x + threadIdx.x;
  if (i < n) dinv[i] = __frsqrt_rn(fmaxf(deg[i], 1e-12f));
}

__global__ void gcn_edge_kernel(const int* __restrict__ src, const int* __restrict__ dst,
                                const float* __restrict__ h, const float* __restrict__ dinv,
                                float* __restrict__ agg, int E) {
  size_t idx = (size_t)blockIdx.x * blockDim.x + threadIdx.x;
  if (idx >= (size_t)E * OUTF) return;
  int e = (int)(idx >> 6);
  int f = (int)(idx & 63);
  int s = src[e], t = dst[e];
  atomicAdd(&agg[(size_t)t * OUTF + f], h[(size_t)s * OUTF + f] * dinv[s] * dinv[t]);
}

// out = prelu(agg + h*dinv^2 (self loop) + b, a)
__global__ void gcn_finish_kernel(const float* __restrict__ agg, const float* __restrict__ h,
                                  const float* __restrict__ dinv, const float* __restrict__ b,
                                  const float* __restrict__ a, float* __restrict__ out, int n) {
  size_t idx = (size_t)blockIdx.x * blockDim.x + threadIdx.x;
  if (idx >= (size_t)n * OUTF) return;
  int i = (int)(idx >> 6), f = (int)(idx & 63);
  float v = agg[idx] + h[idx] * dinv[i] * dinv[i] + b[f];
  out[idx] = (v >= 0.f) ? v : a[f] * v;
}

__global__ __launch_bounds__(64) void l2norm_kernel(float* __restrict__ x) {
  __shared__ float red[64];
  int n = blockIdx.x, t = threadIdx.x;
  float v = x[(size_t)n * OUTF + t];
  red[t] = v * v;
  __syncthreads();
  for (int o = 32; o > 0; o >>= 1) {
    if (t < o) red[t] += red[t + o];
    __syncthreads();
  }
  float nrm = fmaxf(sqrtf(red[0]), 1e-12f);
  x[(size_t)n * OUTF + t] = v / nrm;
}

__global__ void combine_kernel(const float* __restrict__ x1, const float* __restrict__ x2,
                               const float* __restrict__ alpha, float* __restrict__ out,
                               size_t total) {
  size_t i = (size_t)blockIdx.x * blockDim.x + threadIdx.x;
  if (i >= total) return;
  float a0 = alpha[0], a1 = alpha[1];
  float mx = fmaxf(a0, a1);
  float e0 = expf(a0 - mx), e1 = expf(a1 - mx);
  float w0 = e0 / (e0 + e1);
  out[i] = x1[i] * w0 + x2[i] * (1.f - w0);
}

__global__ void gather_rows_kernel(const float* __restrict__ x, const int* __restrict__ perm,
                                   float* __restrict__ out, size_t total) {
  size_t i = (size_t)blockIdx.x * blockDim.x + threadIdx.x;
  if (i >= total) return;
  size_t r = i >> 6, f = i & 63;
  out[i] = x[(size_t)perm[r] * OUTF + f];
}

// ------------------------- discriminator -----------------------------------

__global__ __launch_bounds__(256) void colmean_sigmoid_kernel(
    const float* __restrict__ pos, float* __restrict__ s, int n) {
  __shared__ float red[256];
  int f = blockIdx.x;
  float a = 0.f;
  for (int i = threadIdx.x; i < n; i += 256) a += pos[(size_t)i * OUTF + f];
  red[threadIdx.x] = a;
  __syncthreads();
  for (int o = 128; o > 0; o >>= 1) {
    if (threadIdx.x < o) red[threadIdx.x] += red[threadIdx.x + o];
    __syncthreads();
  }
  if (threadIdx.x == 0) s[f] = 1.f / (1.f + expf(-red[0] / (float)n));
}

__global__ __launch_bounds__(64) void matvec64_kernel(
    const float* __restrict__ W, const float* __restrict__ s, float* __restrict__ ws) {
  int r = threadIdx.x;
  float a = 0.f;
  for (int c = 0; c < OUTF; ++c) a += W[r * OUTF + c] * s[c];
  ws[r] = a;
}

__global__ __launch_bounds__(64) void disc_kernel(
    const float* __restrict__ pos, const float* __restrict__ neg,
    const float* __restrict__ ws, float* __restrict__ acc) {
  __shared__ float red[64];
  int n = blockIdx.x, t = threadIdx.x;
  red[t] = pos[(size_t)n * OUTF + t] * ws[t];
  __syncthreads();
  for (int o = 32; o > 0; o >>= 1) {
    if (t < o) red[t] += red[t + o];
    __syncthreads();
  }
  float dp = red[0];
  __syncthreads();
  red[t] = neg[(size_t)n * OUTF + t] * ws[t];
  __syncthreads();
  for (int o = 32; o > 0; o >>= 1) {
    if (t < o) red[t] += red[t + o];
    __syncthreads();
  }
  float dn = red[0];
  if (t == 0) {
    float sp = 1.f / (1.f + expf(-dp));
    float sn = 1.f / (1.f + expf(-dn));
    atomicAdd(&acc[0], logf(sp + EPSC));
    atomicAdd(&acc[1], logf(1.f - sn + EPSC));
  }
}

__global__ void ctr_final_kernel(const float* __restrict__ acc, float* __restrict__ out) {
  if (threadIdx.x == 0 && blockIdx.x == 0)
    out[0] = -(acc[0] / (float)NR) - (acc[1] / (float)NR);
}

// ------------------------- SpMM decoder ------------------------------------

__global__ void spmm_kernel(const int* __restrict__ rows, const int* __restrict__ cols,
                            const float* __restrict__ vals, const float* __restrict__ x,
                            float* __restrict__ out, int E, int F) {
  size_t idx = (size_t)blockIdx.x * blockDim.x + threadIdx.x;
  if (idx >= (size_t)E * F) return;
  int e = (int)(idx / F);
  int f = (int)(idx % F);
  atomicAdd(&out[(size_t)rows[e] * F + f], vals[e] * x[(size_t)cols[e] * F + f]);
}

// ---------------------------------------------------------------------------

static inline unsigned cdiv(size_t a, size_t b) { return (unsigned)((a + b - 1) / b); }

extern "C" void kernel_launch(void* const* d_in, const int* in_sizes, int n_in,
                              void* d_out, int out_size, void* d_ws, size_t ws_size,
                              hipStream_t stream) {
  (void)n_in; (void)out_size; (void)ws_size;

  // --- input indices (setup_inputs dict order, nested dicts flattened) ---
  const float* x1        = (const float*)d_in[0];
  const float* x2        = (const float*)d_in[1];
  const int*   adj1_idx  = (const int*)d_in[2];
  const float* adj1_vals = (const float*)d_in[3];
  const int*   adj2_idx  = (const int*)d_in[4];
  const float* adj2_vals = (const float*)d_in[5];
  const int*   sp_ei     = (const int*)d_in[6];
  void* const* enc1p     = d_in + 7;    // 14 entries
  void* const* enc2p     = d_in + 21;   // 14 entries
  void* const* fusp      = d_in + 35;   // 13 entries
  const float* dec1_W    = (const float*)d_in[48];
  const float* dec2_W    = (const float*)d_in[49];
  const float* dis_w     = (const float*)d_in[50];
  const float* projs1    = (const float*)d_in[51];
  const float* projs2    = (const float*)d_in[52];
  const float* gum1      = (const float*)d_in[53];
  const float* gum2      = (const float*)d_in[54];
  const int*   perm1     = (const int*)d_in[55];
  const int*   perm2     = (const int*)d_in[56];

  const int E1  = in_sizes[3];
  const int E2  = in_sizes[5];
  const int Esp = in_sizes[6] / 2;

  // --- output layout: dec1 | dec2 | pos | ctr (fp32) ---
  float* out_dec1 = (float*)d_out;
  float* out_dec2 = out_dec1 + (size_t)NR * IN1F;
  float* out_pos  = out_dec2 + (size_t)NR * IN2F;
  float* out_ctr  = out_pos + (size_t)NR * OUTF;

  // --- workspace cursor (allocations rounded to 64 floats for 16B align) ---
  float* wsp = (float*)d_ws;
  auto walloc = [&](size_t nfloats) {
    float* p = wsp;
    wsp += (nfloats + 63) & ~(size_t)63;
    return p;
  };
  float* zcat   = walloc((size_t)NR * 512);   // [layers0..3] concat
  float* t0     = walloc((size_t)NR * HIDN);
  float* qbuf   = walloc((size_t)NR * HIDN);
  float* kbuf   = walloc((size_t)NR * HIDN);
  float* vbuf   = walloc((size_t)NR * HIDN);
  float* qp     = walloc((size_t)NR * MFEAT);
  float* kp     = walloc((size_t)NR * MFEAT);
  float* diagb  = walloc(NR);
  float* rowmax = walloc(NR);
  float* gmax   = walloc(1);
  float* kvsb   = walloc((size_t)KSAMP * MFEAT * HIDN);
  float* kgsum  = walloc((size_t)KSAMP * MFEAT);
  float* znb    = walloc((size_t)NR * HIDN);
  float* e1     = walloc((size_t)NR * OUTF);
  float* e2     = walloc((size_t)NR * OUTF);
  float* e1p    = walloc((size_t)NR * OUTF);
  float* e2p    = walloc((size_t)NR * OUTF);
  float* degb   = walloc(NR);
  float* dinvb  = walloc(NR);
  float* hbuf   = walloc((size_t)NR * OUTF);
  float* aggbuf = walloc((size_t)NR * OUTF);
  float* t1b    = walloc((size_t)NR * OUTF);
  float* t2b    = walloc((size_t)NR * OUTF);
  float* negb   = walloc((size_t)NR * OUTF);
  float* sbuf   = walloc(OUTF);
  float* wsv    = walloc(OUTF);
  float* acc2   = walloc(2);
  float* y1     = walloc((size_t)NR * IN1F);
  float* y2     = walloc((size_t)NR * IN2F);

  auto fillk = [&](float* p, float v, size_t n) {
    fill_kernel<<<cdiv(n, 256), 256, 0, stream>>>(p, v, n);
  };
  auto gemm = [&](const float* A, int lda, const float* B, int ldb, const float* bias,
                  float* C, int ldc, int Md, int Nd, int Kd) {
    dim3 g(cdiv(Nd, BN), cdiv(Md, BM));
    gemm_wmma_bf16<<<g, 256, 0, stream>>>(A, lda, B, ldb, bias, C, ldc, Md, Nd, Kd);
  };

  const float scale = 0.29730177875068026f;  // 128^-0.25, TAU=1

  // --------------------------- MSLA encoder ---------------------------
  auto run_msla = [&](const float* x, int in_feat, void* const* enc,
                      const float* projs, const float* gums, float* eout) {
    const float* fc0_w = (const float*)enc[0];
    const float* fc0_b = (const float*)enc[1];
    const float* fc1_w = (const float*)enc[2];
    const float* fc1_b = (const float*)enc[3];
    const float* ln_g  = (const float*)enc[4];
    const float* ln_b  = (const float*)enc[5];
    const float* wq    = (const float*)enc[6];
    const float* bq    = (const float*)enc[7];
    const float* wk    = (const float*)enc[8];
    const float* bk    = (const float*)enc[9];
    const float* wv    = (const float*)enc[10];
    const float* bv    = (const float*)enc[11];
    const float* wo    = (const float*)enc[12];
    const float* bo    = (const float*)enc[13];

    gemm(x, in_feat, fc0_w, HIDN, fc0_b, t0, HIDN, NR, HIDN, in_feat);
    ln_elu_kernel<<<NR, 128, 0, stream>>>(t0, HIDN, nullptr, 0, ln_g, ln_b, zcat, 512);

    for (int i = 0; i < LNUM; ++i) {
      const float* zprev = zcat + (size_t)i * HIDN;  // ld 512
      const float* P = projs + (size_t)i * MFEAT * HIDN;
      const float* g = gums + (size_t)i * NR * KSAMP;

      gemm(zprev, 512, wq + (size_t)i * HIDN * HIDN, HIDN, bq + (size_t)i * HIDN,
           qbuf, HIDN, NR, HIDN, HIDN);
      gemm(zprev, 512, wk + (size_t)i * HIDN * HIDN, HIDN, bk + (size_t)i * HIDN,
           kbuf, HIDN, NR, HIDN, HIDN);
      gemm(zprev, 512, wv + (size_t)i * HIDN * HIDN, HIDN, bv + (size_t)i * HIDN,
           vbuf, HIDN, NR, HIDN, HIDN);

      // phi(q): per-row stabilization
      favor_dd_kernel<<<NR, 32, 0, stream>>>(qbuf, HIDN, P, qp, diagb, rowmax, scale);
      favor_exp_kernel<<<cdiv((size_t)NR * MFEAT, 256), 256, 0, stream>>>(
          qp, diagb, rowmax, gmax, 1, NR * MFEAT);
      // phi(k): global stabilization
      favor_dd_kernel<<<NR, 32, 0, stream>>>(kbuf, HIDN, P, kp, diagb, rowmax, scale);
      reduce_max_kernel<<<1, 256, 0, stream>>>(rowmax, gmax, NR);
      favor_exp_kernel<<<cdiv((size_t)NR * MFEAT, 256), 256, 0, stream>>>(
          kp, diagb, rowmax, gmax, 0, NR * MFEAT);

      fillk(kvsb, 0.f, (size_t)KSAMP * MFEAT * HIDN);
      fillk(kgsum, 0.f, (size_t)KSAMP * MFEAT);
      kvs_kernel<<<dim3(KSAMP * MFEAT, 20), 128, 0, stream>>>(kp, g, vbuf, HIDN, kvsb, kgsum, NR);
      zn_kernel<<<NR, 128, 0, stream>>>(qp, kvsb, kgsum, znb);

      gemm(znb, HIDN, wo + (size_t)i * HIDN * HIDN, HIDN, bo + (size_t)i * HIDN,
           t0, HIDN, NR, HIDN, HIDN);
      // residual add + LN + ELU, write into next concat slice
      ln_elu_kernel<<<NR, 128, 0, stream>>>(t0, HIDN, zprev, 512,
                                            ln_g + (size_t)(i + 1) * HIDN,
                                            ln_b + (size_t)(i + 1) * HIDN,
                                            zcat + (size_t)(i + 1) * HIDN, 512);
    }
    gemm(zcat, 512, fc1_w, OUTF, fc1_b, eout, OUTF, NR, OUTF, 512);
  };

  run_msla(x1, IN1F, enc1p, projs1, gum1, e1);
  run_msla(x2, IN2F, enc2p, projs2, gum2, e2);

  // --------------------------- fusion (GCN) ---------------------------
  const int* src = sp_ei;
  const int* dst = sp_ei + Esp;
  fillk(degb, 1.f, NR);  // self-loops
  deg_edge_kernel<<<cdiv(Esp, 256), 256, 0, stream>>>(dst, degb, Esp);
  dinv_kernel<<<cdiv(NR, 256), 256, 0, stream>>>(degb, dinvb, NR);

  const float* fw1 = (const float*)fusp[0];  const float* fb1 = (const float*)fusp[1];
  const float* fw2 = (const float*)fusp[2];  const float* fb2 = (const float*)fusp[3];
  const float* fw3 = (const float*)fusp[4];  const float* fb3 = (const float*)fusp[5];
  const float* fw4 = (const float*)fusp[6];  const float* fb4 = (const float*)fusp[7];
  const float* fa1 = (const float*)fusp[8];  const float* fa2 = (const float*)fusp[9];
  const float* fa3 = (const float*)fusp[10]; const float* fa4 = (const float*)fusp[11];
  const float* falpha = (const float*)fusp[12];

  auto gcn = [&](const float* xin, const float* W, const float* b, const float* a, float* out) {
    gemm(xin, OUTF, W, OUTF, nullptr, hbuf, OUTF, NR, OUTF, OUTF);
    fillk(aggbuf, 0.f, (size_t)NR * OUTF);
    gcn_edge_kernel<<<cdiv((size_t)Esp * OUTF, 256), 256, 0, stream>>>(src, dst, hbuf, dinvb, aggbuf, Esp);
    gcn_finish_kernel<<<cdiv((size_t)NR * OUTF, 256), 256, 0, stream>>>(aggbuf, hbuf, dinvb, b, a, out, NR);
  };
  auto fusion = [&](const float* f1, const float* f2, float* out) {
    gcn(f1, fw1, fb1, fa1, t1b);
    gcn(t1b, fw3, fb3, fa2, t1b);
    l2norm_kernel<<<NR, 64, 0, stream>>>(t1b);
    gcn(f2, fw2, fb2, fa3, t2b);
    gcn(t2b, fw4, fb4, fa4, t2b);
    l2norm_kernel<<<NR, 64, 0, stream>>>(t2b);
    combine_kernel<<<cdiv((size_t)NR * OUTF, 256), 256, 0, stream>>>(t1b, t2b, falpha, out, (size_t)NR * OUTF);
  };

  fusion(e1, e2, out_pos);
  gather_rows_kernel<<<cdiv((size_t)NR * OUTF, 256), 256, 0, stream>>>(e1, perm1, e1p, (size_t)NR * OUTF);
  gather_rows_kernel<<<cdiv((size_t)NR * OUTF, 256), 256, 0, stream>>>(e2, perm2, e2p, (size_t)NR * OUTF);
  fusion(e1p, e2p, negb);

  // --------------------------- discriminator --------------------------
  colmean_sigmoid_kernel<<<OUTF, 256, 0, stream>>>(out_pos, sbuf, NR);
  matvec64_kernel<<<1, 64, 0, stream>>>(dis_w, sbuf, wsv);
  fillk(acc2, 0.f, 2);
  disc_kernel<<<NR, 64, 0, stream>>>(out_pos, negb, wsv, acc2);
  ctr_final_kernel<<<1, 1, 0, stream>>>(acc2, out_ctr);

  // --------------------------- decoders (GEMM + SpMM) -----------------
  gemm(out_pos, OUTF, dec1_W, IN1F, nullptr, y1, IN1F, NR, IN1F, OUTF);
  gemm(out_pos, OUTF, dec2_W, IN2F, nullptr, y2, IN2F, NR, IN2F, OUTF);
  fillk(out_dec1, 0.f, (size_t)NR * IN1F);
  fillk(out_dec2, 0.f, (size_t)NR * IN2F);
  spmm_kernel<<<cdiv((size_t)E1 * IN1F, 256), 256, 0, stream>>>(
      adj1_idx, adj1_idx + E1, adj1_vals, y1, out_dec1, E1, IN1F);
  spmm_kernel<<<cdiv((size_t)E2 * IN2F, 256), 256, 0, stream>>>(
      adj2_idx, adj2_idx + E2, adj2_vals, y2, out_dec2, E2, IN2F);
}
